// FeatherNet_27023934227230
// MI455X (gfx1250) — matrix operand order repaired and checked by reading, not compiled, not run on hardware
//
#include <hip/hip_runtime.h>

// ---------------------------------------------------------------------------
// Types for CDNA5 WMMA (wave32, 16x16x32 bf16 -> f32)
// ---------------------------------------------------------------------------
typedef __bf16 bf16_t;
typedef __attribute__((ext_vector_type(16))) __bf16        v16bf;
typedef __attribute__((ext_vector_type(8)))  float         v8f;
typedef __attribute__((ext_vector_type(4)))  unsigned int  u32x4;
typedef __attribute__((ext_vector_type(4)))  int           v4i;

#define WMMA_BF16(a, b, c) \
  __builtin_amdgcn_wmma_f32_16x16x32_bf16(false, (a), false, (b), (short)0, (c), false, false)

// CDNA5 async global->LDS path (ASYNCcnt); falls back to register staging if
// the toolchain doesn't expose the builtin.
#if defined(__gfx1250__) && __has_builtin(__builtin_amdgcn_global_load_async_to_lds_b128)
#define HAVE_ASYNC_LDS 1
#else
#define HAVE_ASYNC_LDS 0
#endif

#if HAVE_ASYNC_LDS
// builtin signature is (int4 AS1*, int4 AS3*, imm offset, imm cpol)
typedef __attribute__((address_space(1))) v4i gv4i;
typedef __attribute__((address_space(3))) v4i lv4i;
__device__ __forceinline__ void asyncCopyB128(const bf16_t* g, bf16_t* l) {
  __builtin_amdgcn_global_load_async_to_lds_b128(
      (gv4i*)const_cast<bf16_t*>(g), (lv4i*)l, 0, 0);
}
__device__ __forceinline__ void waitAsync() {
#if __has_builtin(__builtin_amdgcn_s_wait_asynccnt)
  __builtin_amdgcn_s_wait_asynccnt(0);
#else
  asm volatile("s_wait_asynccnt 0x0" ::: "memory");
#endif
}
#endif

// ---------------------------------------------------------------------------
// Problem dimensions
// ---------------------------------------------------------------------------
constexpr int  kB   = 256;    // batch
constexpr int  kIN  = 1024;
constexpr int  kH   = 4096;
constexpr int  kOUT = 1024;
constexpr long kSN  = 5018;   // SIZE_N
constexpr long kSM  = 1255;   // SIZE_M
constexpr int  kPA  = 5120;   // SIZE_N padded so 128x64 block tiles divide evenly
constexpr int  kPK  = 1280;   // SIZE_M padded to 32 (40 K-steps)

// flat parameter offsets (elements), module order: W1 b1 W2 b2 W3 b3
constexpr long OFF_W1 = 0;
constexpr long OFF_B1 = (long)kH * kIN;            //  4,194,304
constexpr long OFF_W2 = OFF_B1 + kH;               //  4,198,400
constexpr long OFF_B2 = OFF_W2 + (long)kH * kH;    // 20,975,616
constexpr long OFF_W3 = OFF_B2 + kH;               // 20,979,712
constexpr long OFF_B3 = OFF_W3 + (long)kOUT * kH;  // 25,174,016

// synthesis GEMM block tiling
constexpr int BM = 128;          // block tile M
constexpr int BN = 64;           // block tile N
constexpr int BK = 32;           // K step (bf16 WMMA depth)
constexpr int LDSS = 40;         // LDS row stride (32 + 8 pad -> 20 banks, conflict-free frags)
constexpr int KSTEPS = kPK / BK; // 40
constexpr int NBUF = HAVE_ASYNC_LDS ? 2 : 1;   // LDS double-buffer only on async path

// ---------------------------------------------------------------------------
// Fragment loaders (16B loads; global -> global_load_b128, LDS -> ds_load_b128)
// A fragment (16x32 bf16, MxK): lane row = lane&15; lanes<16 hold K 0..7 &
// 16..23, lanes>=16 hold K 8..15 & 24..31  (ISA 7.12.2 16-bit A layout).
// B fragment (32x16 bf16, KxN) from B^T rows: lane col = lane&15; lanes<16
// hold K 0..7 & 16..23, lanes>=16 hold K 8..15 & 24..31 (same as A).
// ---------------------------------------------------------------------------
__device__ __forceinline__ v16bf loadFragA(const bf16_t* rowPtr, int k0, int lane) {
  const int cb = (lane >> 4) << 3;  // 0 or 8
  union { v16bf v; u32x4 q[2]; } u;
  u.q[0] = *reinterpret_cast<const u32x4*>(rowPtr + k0 + cb);
  u.q[1] = *reinterpret_cast<const u32x4*>(rowPtr + k0 + 16 + cb);
  return u.v;
}

__device__ __forceinline__ v16bf loadFragB(const bf16_t* rowPtr, int k0, int lane) {
  const int cb = (lane >> 4) << 3;  // 0 or 8
  union { v16bf v; u32x4 q[2]; } u;
  u.q[0] = *reinterpret_cast<const u32x4*>(rowPtr + k0 + cb);
  u.q[1] = *reinterpret_cast<const u32x4*>(rowPtr + k0 + 16 + cb);
  return u.v;
}

// 12-WMMA compute step for one 32x32 wave tile (compensated bf16 product)
__device__ __forceinline__ void wmmaStep(const bf16_t* sAh, const bf16_t* sAl,
                                         const bf16_t* sBh, const bf16_t* sBl,
                                         int wm, int wn, int r, int lane,
                                         v8f acc[2][2]) {
  v16bf ah[2], al[2], bh[2], bl[2];
  #pragma unroll
  for (int i = 0; i < 2; ++i) {
    const int row = wm * 32 + 16 * i + r;
    ah[i] = loadFragA(sAh + row * LDSS, 0, lane);
    al[i] = loadFragA(sAl + row * LDSS, 0, lane);
  }
  #pragma unroll
  for (int j = 0; j < 2; ++j) {
    const int row = wn * 32 + 16 * j + r;
    bh[j] = loadFragB(sBh + row * LDSS, 0, lane);
    bl[j] = loadFragB(sBl + row * LDSS, 0, lane);
  }
  #pragma unroll
  for (int i = 0; i < 2; ++i)
    #pragma unroll
    for (int j = 0; j < 2; ++j) {
      acc[i][j] = WMMA_BF16(ah[i], bh[j], acc[i][j]);
      acc[i][j] = WMMA_BF16(ah[i], bl[j], acc[i][j]);
      acc[i][j] = WMMA_BF16(al[i], bh[j], acc[i][j]);
    }
}

// ---------------------------------------------------------------------------
// fp32 -> (bf16 hi, bf16 lo) split/pad conversion kernels
// ---------------------------------------------------------------------------
__global__ void k_split(const float* __restrict__ in,
                        bf16_t* __restrict__ hi, bf16_t* __restrict__ lo,
                        int rows, int cols, int rowsP, int colsP) {
  long i = (long)blockIdx.x * blockDim.x + threadIdx.x;
  long total = (long)rowsP * colsP;
  if (i >= total) return;
  int rr = (int)(i / colsP), cc = (int)(i % colsP);
  float f = (rr < rows && cc < cols) ? in[(long)rr * cols + cc] : 0.0f;
  bf16_t h = (bf16_t)f;
  hi[i] = h;
  lo[i] = (bf16_t)(f - (float)h);
}

// out[r][c] = in[c][r]; in is [inRows, inCols], out is [rowsP, colsP] padded
__global__ void k_split_T(const float* __restrict__ in,
                          bf16_t* __restrict__ hi, bf16_t* __restrict__ lo,
                          int inRows, int inCols, int rowsP, int colsP) {
  long i = (long)blockIdx.x * blockDim.x + threadIdx.x;
  long total = (long)rowsP * colsP;
  if (i >= total) return;
  int rr = (int)(i / colsP), cc = (int)(i % colsP);
  float f = (cc < inRows && rr < inCols) ? in[(long)cc * inCols + rr] : 0.0f;
  bf16_t h = (bf16_t)f;
  hi[i] = h;
  lo[i] = (bf16_t)(f - (float)h);
}

// ---------------------------------------------------------------------------
// Synthesis GEMM: flat = V1 @ V2  (compensated bf16, f32 accumulate)
// LDS-staged: 256 threads (8 waves); block tile 128x64x32; waves 4(M) x 2(N),
// each wave computing a 2x2 block of 16x16 tiles (12 WMMAs / K-step).
// Async path: double-buffered LDS fed by GLOBAL_LOAD_ASYNC_TO_LDS_B128;
// stage s+1 is in flight (ASYNCcnt) while stage s feeds the WMMAs.
// Fallback: register-staged single-buffer pipeline.
// ---------------------------------------------------------------------------
__global__ __launch_bounds__(256) void k_synth(
    const bf16_t* __restrict__ Ahi, const bf16_t* __restrict__ Alo,
    const bf16_t* __restrict__ Bhi, const bf16_t* __restrict__ Blo,
    bf16_t* __restrict__ Fhi, bf16_t* __restrict__ Flo) {
  __shared__ bf16_t sAh[NBUF][BM * LDSS];
  __shared__ bf16_t sAl[NBUF][BM * LDSS];
  __shared__ bf16_t sBh[NBUF][BN * LDSS];
  __shared__ bf16_t sBl[NBUF][BN * LDSS];

  const int tid  = threadIdx.x;
  const int lane = tid & 31;
  const int wave = tid >> 5;
  const int wm = wave >> 1;           // 0..3
  const int wn = wave & 1;            // 0..1
  const int r  = lane & 15;

  const long mBase = (long)blockIdx.y * BM;
  const long nBase = (long)blockIdx.x * BN;

  // cooperative-load coordinates: A tile = 128 rows x 4 chunks of 8 bf16;
  // each thread owns A chunks {tid, tid+256} and B chunk {tid}
  const int ar0 = tid >> 2,         aq0 = tid & 3;
  const int ar1 = (tid + 256) >> 2, aq1 = (tid + 256) & 3;
  const int br  = tid >> 2,         bq  = tid & 3;

  const bf16_t* gAh0 = Ahi + (mBase + ar0) * kPK + aq0 * 8;
  const bf16_t* gAh1 = Ahi + (mBase + ar1) * kPK + aq1 * 8;
  const bf16_t* gAl0 = Alo + (mBase + ar0) * kPK + aq0 * 8;
  const bf16_t* gAl1 = Alo + (mBase + ar1) * kPK + aq1 * 8;
  const bf16_t* gBh  = Bhi + (nBase + br) * kPK + bq * 8;
  const bf16_t* gBl  = Blo + (nBase + br) * kPK + bq * 8;

  v8f acc[2][2] = {};

#if HAVE_ASYNC_LDS
  auto issueStage = [&](int s, int b) {
    const int ko = s * BK;
    asyncCopyB128(gAh0 + ko, &sAh[b][ar0 * LDSS + aq0 * 8]);
    asyncCopyB128(gAh1 + ko, &sAh[b][ar1 * LDSS + aq1 * 8]);
    asyncCopyB128(gAl0 + ko, &sAl[b][ar0 * LDSS + aq0 * 8]);
    asyncCopyB128(gAl1 + ko, &sAl[b][ar1 * LDSS + aq1 * 8]);
    asyncCopyB128(gBh  + ko, &sBh[b][br * LDSS + bq * 8]);
    asyncCopyB128(gBl  + ko, &sBl[b][br * LDSS + bq * 8]);
  };

  issueStage(0, 0);
  for (int s = 0; s < KSTEPS; ++s) {
    const int cur = s & 1;
    // only stage-s loads are in flight here; wait for our wave's copies,
    // then barrier to publish every wave's LDS writes (and to guarantee all
    // waves finished reading buf[cur^1] before we overwrite it below).
    waitAsync();
    __syncthreads();
    if (s + 1 < KSTEPS) issueStage(s + 1, cur ^ 1);  // flies under the WMMAs
    wmmaStep(sAh[cur], sAl[cur], sBh[cur], sBl[cur], wm, wn, r, lane, acc);
  }
#else
  // prologue: global loads for K-step 0
  u32x4 ra0 = *reinterpret_cast<const u32x4*>(gAh0);
  u32x4 ra1 = *reinterpret_cast<const u32x4*>(gAh1);
  u32x4 rl0 = *reinterpret_cast<const u32x4*>(gAl0);
  u32x4 rl1 = *reinterpret_cast<const u32x4*>(gAl1);
  u32x4 rb  = *reinterpret_cast<const u32x4*>(gBh);
  u32x4 rc  = *reinterpret_cast<const u32x4*>(gBl);

  for (int s = 0; s < KSTEPS; ++s) {
    *reinterpret_cast<u32x4*>(&sAh[0][ar0 * LDSS + aq0 * 8]) = ra0;
    *reinterpret_cast<u32x4*>(&sAh[0][ar1 * LDSS + aq1 * 8]) = ra1;
    *reinterpret_cast<u32x4*>(&sAl[0][ar0 * LDSS + aq0 * 8]) = rl0;
    *reinterpret_cast<u32x4*>(&sAl[0][ar1 * LDSS + aq1 * 8]) = rl1;
    *reinterpret_cast<u32x4*>(&sBh[0][br  * LDSS + bq  * 8]) = rb;
    *reinterpret_cast<u32x4*>(&sBl[0][br  * LDSS + bq  * 8]) = rc;
    __syncthreads();

    if (s + 1 < KSTEPS) {      // issue next K-step's loads under the WMMAs
      const int ko = (s + 1) * BK;
      ra0 = *reinterpret_cast<const u32x4*>(gAh0 + ko);
      ra1 = *reinterpret_cast<const u32x4*>(gAh1 + ko);
      rl0 = *reinterpret_cast<const u32x4*>(gAl0 + ko);
      rl1 = *reinterpret_cast<const u32x4*>(gAl1 + ko);
      rb  = *reinterpret_cast<const u32x4*>(gBh + ko);
      rc  = *reinterpret_cast<const u32x4*>(gBl + ko);
    }

    wmmaStep(sAh[0], sAl[0], sBh[0], sBl[0], wm, wn, r, lane, acc);
    __syncthreads();
  }
#endif

  // C layout: element (m, n) -> vgpr v = m&7, lane = (m>=8)*16 + n
  const int nl = lane & 15, mh = (lane >> 4) << 3;
  #pragma unroll
  for (int i = 0; i < 2; ++i)
    #pragma unroll
    for (int j = 0; j < 2; ++j) {
      long col = nBase + wn * 32 + 16 * j + nl;
      if (col >= kSN) continue;
      union { v8f v; float f[8]; } a; a.v = acc[i][j];
      #pragma unroll
      for (int v = 0; v < 8; ++v) {
        long row = mBase + wm * 32 + 16 * i + mh + v;
        if (row < kSN) {
          long idx = row * kSN + col;
          float f = a.f[v];
          bf16_t h = (bf16_t)f;
          Fhi[idx] = h;
          Flo[idx] = (bf16_t)(f - (float)h);
        }
      }
    }
}

// ---------------------------------------------------------------------------
// MLP layer GEMM: Y = epi( wScale * (X @ W^T) + bScale * bias )
// X  (hi/lo)  [M x K] row-major;  W (hi/lo) [N x K] row-major (slice of flat,
// which IS the B-transposed layout WMMA wants).  One 16x16 tile per wave.
// LAST=false: ReLU, re-split to bf16 hi/lo.   LAST=true: fp32 output.
// ---------------------------------------------------------------------------
template <bool LAST>
__global__ __launch_bounds__(256) void k_mlp(
    const bf16_t* __restrict__ Xhi, const bf16_t* __restrict__ Xlo,
    const bf16_t* __restrict__ Whi, const bf16_t* __restrict__ Wlo,
    const bf16_t* __restrict__ Bhi, const bf16_t* __restrict__ Blo,
    const float* __restrict__ wScale, const float* __restrict__ bScale,
    int M, int N, int K,
    bf16_t* __restrict__ Yhi, bf16_t* __restrict__ Ylo,
    float* __restrict__ Yf) {
  const int lane = threadIdx.x & 31;
  const int wave = threadIdx.x >> 5;
  const int tilesN = N >> 4;
  int t = blockIdx.x * (blockDim.x >> 5) + wave;
  if (t >= (M >> 4) * tilesN) return;
  const int tm = t / tilesN, tn = t % tilesN;
  const int r = lane & 15;

  const bf16_t* xh = Xhi + (long)(tm * 16 + r) * K;
  const bf16_t* xl = Xlo + (long)(tm * 16 + r) * K;
  const bf16_t* wh = Whi + (long)(tn * 16 + r) * K;
  const bf16_t* wl = Wlo + (long)(tn * 16 + r) * K;

  v8f acc = {};
  for (int k0 = 0; k0 < K; k0 += 32) {
    if (k0 + 64 < K) {  // pull the next weight lines toward the WGP
      __builtin_prefetch(wh + k0 + 64, 0, 1);
      __builtin_prefetch(wl + k0 + 64, 0, 1);
    }
    v16bf ah = loadFragA(xh, k0, lane), al = loadFragA(xl, k0, lane);
    v16bf bh = loadFragB(wh, k0, lane), bl = loadFragB(wl, k0, lane);
    acc = WMMA_BF16(ah, bh, acc);
    acc = WMMA_BF16(ah, bl, acc);
    acc = WMMA_BF16(al, bh, acc);
  }

  const float ws = wScale[0], bs = bScale[0];
  const int nl = lane & 15, mh = (lane >> 4) << 3;
  const int col = tn * 16 + nl;
  const float bias = bs * ((float)Bhi[col] + (float)Blo[col]);
  union { v8f v; float f[8]; } a; a.v = acc;
  #pragma unroll
  for (int v = 0; v < 8; ++v) {
    int row = tm * 16 + mh + v;
    float y = ws * a.f[v] + bias;
    long idx = (long)row * N + col;
    if (LAST) {
      Yf[idx] = y;
    } else {
      y = fmaxf(y, 0.0f);
      bf16_t h = (bf16_t)y;
      Yhi[idx] = h;
      Ylo[idx] = (bf16_t)(y - (float)h);
    }
  }
}

// ---------------------------------------------------------------------------
// Host launcher
// ---------------------------------------------------------------------------
extern "C" void kernel_launch(void* const* d_in, const int* in_sizes, int n_in,
                              void* d_out, int out_size, void* d_ws, size_t ws_size,
                              hipStream_t stream) {
  (void)in_sizes; (void)n_in; (void)out_size; (void)ws_size;
  const float* x   = (const float*)d_in[0];
  const float* V1  = (const float*)d_in[1];
  const float* V2  = (const float*)d_in[2];
  const float* w1p = (const float*)d_in[3];
  const float* b1p = (const float*)d_in[4];
  const float* w2p = (const float*)d_in[5];
  const float* b2p = (const float*)d_in[6];
  const float* w3p = (const float*)d_in[7];
  const float* b3p = (const float*)d_in[8];
  float* out = (float*)d_out;

  // workspace carve-out (256B aligned)
  char* ws = (char*)d_ws;
  size_t off = 0;
  auto carve = [&](size_t bytes) {
    size_t o = off;
    off = (off + bytes + 255) & ~(size_t)255;
    return o;
  };
  bf16_t* pFhi  = (bf16_t*)(ws + carve((size_t)kSN * kSN * 2));
  bf16_t* pFlo  = (bf16_t*)(ws + carve((size_t)kSN * kSN * 2));
  bf16_t* pAhi  = (bf16_t*)(ws + carve((size_t)kPA * kPK * 2));
  bf16_t* pAlo  = (bf16_t*)(ws + carve((size_t)kPA * kPK * 2));
  bf16_t* pBhi  = (bf16_t*)(ws + carve((size_t)kPA * kPK * 2));
  bf16_t* pBlo  = (bf16_t*)(ws + carve((size_t)kPA * kPK * 2));
  bf16_t* pXhi  = (bf16_t*)(ws + carve((size_t)kB * kIN * 2));
  bf16_t* pXlo  = (bf16_t*)(ws + carve((size_t)kB * kIN * 2));
  bf16_t* pH1hi = (bf16_t*)(ws + carve((size_t)kB * kH * 2));
  bf16_t* pH1lo = (bf16_t*)(ws + carve((size_t)kB * kH * 2));
  bf16_t* pH2hi = (bf16_t*)(ws + carve((size_t)kB * kH * 2));
  bf16_t* pH2lo = (bf16_t*)(ws + carve((size_t)kB * kH * 2));

  const int blk = 256;

  // 1) fp32 -> bf16 hi/lo splits (with padding; V2 is transposed to K-major)
  {
    long tot = (long)kPA * kPK;
    int g = (int)((tot + blk - 1) / blk);
    k_split  <<<g, blk, 0, stream>>>(V1, pAhi, pAlo, (int)kSN, (int)kSM, kPA, kPK);
    k_split_T<<<g, blk, 0, stream>>>(V2, pBhi, pBlo, (int)kSM, (int)kSN, kPA, kPK);
  }
  {
    long tot = (long)kB * kIN;
    int g = (int)((tot + blk - 1) / blk);
    k_split<<<g, blk, 0, stream>>>(x, pXhi, pXlo, kB, kIN, kB, kIN);
  }

  // 2) weight synthesis GEMM: flat = V1 @ V2 (LDS-staged compensated bf16 WMMA)
  {
    dim3 grid(kPA / BN, kPA / BM);   // (80, 40)
    k_synth<<<grid, blk, 0, stream>>>(pAhi, pAlo, pBhi, pBlo, pFhi, pFlo);
  }

  // 3) MLP layers (bias slices of flat; weight scale folded into epilogue)
  {
    int t = (kB / 16) * (kH / 16);
    int g = (t + 7) / 8;
    k_mlp<false><<<g, blk, 0, stream>>>(pXhi, pXlo,
                                        pFhi + OFF_W1, pFlo + OFF_W1,
                                        pFhi + OFF_B1, pFlo + OFF_B1,
                                        w1p, b1p, kB, kH, kIN,
                                        pH1hi, pH1lo, nullptr);
    k_mlp<false><<<g, blk, 0, stream>>>(pH1hi, pH1lo,
                                        pFhi + OFF_W2, pFlo + OFF_W2,
                                        pFhi + OFF_B2, pFlo + OFF_B2,
                                        w2p, b2p, kB, kH, kH,
                                        pH2hi, pH2lo, nullptr);
  }
  {
    int t = (kB / 16) * (kOUT / 16);
    int g = (t + 7) / 8;
    k_mlp<true><<<g, blk, 0, stream>>>(pH2hi, pH2lo,
                                       pFhi + OFF_W3, pFlo + OFF_W3,
                                       pFhi + OFF_B3, pFlo + OFF_B3,
                                       w3p, b3p, kB, kOUT, kH,
                                       nullptr, nullptr, out);
  }
}
